// TransformerBlock_24120536334932
// MI455X (gfx1250) — matrix operand order repaired and checked by reading, not compiled, not run on hardware
//
#include <hip/hip_runtime.h>
#include <hip/hip_bf16.h>

// ---------------------------------------------------------------------------
// Types for CDNA5 WMMA (wave32, 16x16x32 bf16 -> f32)
// ---------------------------------------------------------------------------
typedef __attribute__((ext_vector_type(16))) __bf16 v16bf;
typedef __attribute__((ext_vector_type(8)))  __bf16 v8bf;
typedef __attribute__((ext_vector_type(4)))  __bf16 v4bf;
typedef __attribute__((ext_vector_type(8)))  float  v8f;

#define LDA  72   // LDS row stride (bf16) for 128x64 A tiles: 144B, 16B-aligned, bank-spread
#define LDWT 72   // LDS column stride (bf16) for transposed 64x64 weights

__device__ inline v16bf concat8(v8bf lo, v8bf hi) {
  return __builtin_shufflevector(lo, hi, 0, 1, 2, 3, 4, 5, 6, 7,
                                 8, 9, 10, 11, 12, 13, 14, 15);
}

// ---------------------------------------------------------------------------
// WMMA core: one wave computes a 16-row x 64-col tile, K=64 (2 steps of K=32).
// A tile in LDS row-major (stride LDA).  Weights in LDS TRANSPOSED:
// WsT[c*LDWT + k] so each lane's B fragment is two 16-byte runs along k
// (two ds_load_b128).  Fragment layouts per ISA 7.12.2.
// ---------------------------------------------------------------------------
__device__ inline void wmma_rowtile_64(const __bf16* As, const __bf16* WsT,
                                       int rowTile, v8f acc[4]) {
  const int lane = threadIdx.x & 31;
  const int half = lane >> 4;
  const int lm   = lane & 15;

  v16bf af[2];
#pragma unroll
  for (int ks = 0; ks < 2; ++ks) {
    const __bf16* arow = As + (rowTile + lm) * LDA + ks * 32;
    const v8bf lo = *(const v8bf*)(arow + half * 8);
    const v8bf hi = *(const v8bf*)(arow + 16 + half * 8);
    af[ks] = concat8(lo, hi);
  }

#pragma unroll
  for (int ct = 0; ct < 4; ++ct) {
    v8f a = {};
#pragma unroll
    for (int ks = 0; ks < 2; ++ks) {
      const __bf16* bcol = WsT + (ct * 16 + lm) * LDWT + ks * 32;
      const v8bf lo = *(const v8bf*)(bcol + half * 8);
      const v8bf hi = *(const v8bf*)(bcol + 16 + half * 8);
      a = __builtin_amdgcn_wmma_f32_16x16x32_bf16(false, af[ks], false,
                                                  concat8(lo, hi),
                                                  (short)0, a, false, false);
    }
    acc[ct] = a;
  }
}

// Stage 64x64 f32 weights into LDS transposed bf16 (coalesced global reads).
__device__ inline void stage_weightsT(const float* __restrict__ W, __bf16* WsT,
                                      int t) {
  for (int i = t; i < 64 * 64; i += 256) {
    const int k = i >> 6, c = i & 63;
    WsT[c * LDWT + k] = (__bf16)W[i];
  }
}

// Plain C-store: wave-uniform fast path, compile-time bias/act specialization.
template <bool HAS_BIAS, bool ACT>
__device__ inline void store_rowtile(const v8f acc[4], const float* bs,
                                     float* out, int rowBase, int M) {
  const int t = threadIdx.x;
  const int wave = t >> 5, lane = t & 31, half = lane >> 4, lm = lane & 15;
  const int row0 = rowBase + wave * 16 + half * 8;
  const bool full = (rowBase + wave * 16 + 16) <= M;  // uniform per wave
  if (full) {
#pragma unroll
    for (int ct = 0; ct < 4; ++ct) {
      const int col = ct * 16 + lm;
      const float bcol = HAS_BIAS ? bs[col] : 0.f;
#pragma unroll
      for (int r = 0; r < 8; ++r) {
        float v = acc[ct][r] + bcol;
        if (ACT) v = fmaxf(v, 0.f);
        out[(size_t)(row0 + r) * 64 + col] = v;
      }
    }
  } else {
#pragma unroll
    for (int ct = 0; ct < 4; ++ct) {
      const int col = ct * 16 + lm;
      const float bcol = HAS_BIAS ? bs[col] : 0.f;
#pragma unroll
      for (int r = 0; r < 8; ++r) {
        const int grow = row0 + r;
        if (grow < M) {
          float v = acc[ct][r] + bcol;
          if (ACT) v = fmaxf(v, 0.f);
          out[(size_t)grow * 64 + col] = v;
        }
      }
    }
  }
}

// Edge-GEMM epilogue: bias, store f32, per-channel BN stats.  Lanes lm and
// lm+16 share a column -> pair-reduce with shfl before ds_add_f32.
__device__ inline void edge_epilogue(const v8f acc[4], const float* bs,
                                     float* out, int base, int Etot,
                                     float* ssum, float* ssq) {
  const int t = threadIdx.x;
  const int wave = t >> 5, lane = t & 31, half = lane >> 4, lm = lane & 15;
  const int row0 = base + wave * 16 + half * 8;
  const bool full = (base + wave * 16 + 16) <= Etot;  // uniform per wave
  if (full) {
#pragma unroll
    for (int ct = 0; ct < 4; ++ct) {
      const int col = ct * 16 + lm;
      const float bcol = bs[col];
      float ls = 0.f, lq = 0.f;
#pragma unroll
      for (int r = 0; r < 8; ++r) {
        const float v = acc[ct][r] + bcol;
        out[(size_t)(row0 + r) * 64 + col] = v;
        ls += v;
        lq += v * v;
      }
      ls += __shfl_xor(ls, 16, 32);
      lq += __shfl_xor(lq, 16, 32);
      if (half == 0) {
        atomicAdd(&ssum[col], ls);
        atomicAdd(&ssq[col], lq);
      }
    }
  } else {
#pragma unroll
    for (int ct = 0; ct < 4; ++ct) {
      const int col = ct * 16 + lm;
      const float bcol = bs[col];
      float ls = 0.f, lq = 0.f;
#pragma unroll
      for (int r = 0; r < 8; ++r) {
        const int ge = row0 + r;
        if (ge < Etot) {
          const float v = acc[ct][r] + bcol;
          out[(size_t)ge * 64 + col] = v;
          ls += v;
          lq += v * v;
        }
      }
      ls += __shfl_xor(ls, 16, 32);
      lq += __shfl_xor(lq, 16, 32);
      if (half == 0) {
        atomicAdd(&ssum[col], ls);
        atomicAdd(&ssq[col], lq);
      }
    }
  }
}

// ---------------------------------------------------------------------------
// K0: init workspace (BN stat accumulators, softmax max/sum, output accum)
// ---------------------------------------------------------------------------
__global__ void __launch_bounds__(256)
init_kernel(float* nodesum, int* nodemax, float* out_pre, float* stats, int Nc) {
  const int i = blockIdx.x * 256 + threadIdx.x;
  if (i < Nc) {
    nodesum[i] = 0.f;
    nodemax[i] = (int)0x80000000;  // below any monotone-encoded float
    out_pre[i] = 0.f;
  }
  if (i < 512) stats[i] = 0.f;  // 4 BN layers x (sum[64] + sumsq[64])
}

// ---------------------------------------------------------------------------
// K1: node GEMM  out = act(A[Mx64] @ W[64x64] + b)   (bf16 WMMA, f32 out)
// ---------------------------------------------------------------------------
template <bool HAS_BIAS, bool ACT>
__global__ void __launch_bounds__(256)
gemm64_kernel(const float* __restrict__ A, const float* __restrict__ W,
              const float* __restrict__ bias, float* __restrict__ out, int M) {
  __shared__ __bf16 As[128 * LDA];
  __shared__ __bf16 WsT[64 * LDWT];
  __shared__ float  bs[64];

  const int t = threadIdx.x;
  const int rowBase = blockIdx.x * 128;

  stage_weightsT(W, WsT, t);
  if (HAS_BIAS && t < 64) bs[t] = bias[t];
#pragma unroll
  for (int j = 0; j < 8; ++j) {
    const int idx4 = t + j * 256;      // 2048 float4 groups = 128x64
    const int r = idx4 >> 4;
    const int c4 = (idx4 & 15) * 4;
    const int gr = rowBase + r;
    float4 v = make_float4(0.f, 0.f, 0.f, 0.f);
    if (gr < M) v = *(const float4*)(A + (size_t)gr * 64 + c4);
    v4bf b;
    b[0] = (__bf16)v.x; b[1] = (__bf16)v.y; b[2] = (__bf16)v.z; b[3] = (__bf16)v.w;
    *(v4bf*)(As + r * LDA + c4) = b;
  }
  __syncthreads();

  v8f acc[4];
  wmma_rowtile_64(As, WsT, (t >> 5) * 16, acc);
  store_rowtile<HAS_BIAS, ACT>(acc, bs, out, rowBase, M);
}

// ---------------------------------------------------------------------------
// K1b: fused projection GEMMs: stage h once, multiply by 3 resident weight
// tiles (a_src / a_dst / v).  24 WMMAs per wave, h read from HBM only once.
// ---------------------------------------------------------------------------
__global__ void __launch_bounds__(256)
proj3_kernel(const float* __restrict__ h, const float* __restrict__ W0,
             const float* __restrict__ W1, const float* __restrict__ W2,
             float* __restrict__ o0, float* __restrict__ o1,
             float* __restrict__ o2, int M) {
  __shared__ __bf16 As[128 * LDA];
  __shared__ __bf16 WsT[3][64 * LDWT];

  const int t = threadIdx.x;
  const int rowBase = blockIdx.x * 128;

  stage_weightsT(W0, WsT[0], t);
  stage_weightsT(W1, WsT[1], t);
  stage_weightsT(W2, WsT[2], t);
#pragma unroll
  for (int j = 0; j < 8; ++j) {
    const int idx4 = t + j * 256;
    const int r = idx4 >> 4;
    const int c4 = (idx4 & 15) * 4;
    const int gr = rowBase + r;
    float4 v = make_float4(0.f, 0.f, 0.f, 0.f);
    if (gr < M) v = *(const float4*)(h + (size_t)gr * 64 + c4);
    v4bf b;
    b[0] = (__bf16)v.x; b[1] = (__bf16)v.y; b[2] = (__bf16)v.z; b[3] = (__bf16)v.w;
    *(v4bf*)(As + r * LDA + c4) = b;
  }
  __syncthreads();

  const int rowTile = (t >> 5) * 16;
  v8f acc[4];
  wmma_rowtile_64(As, WsT[0], rowTile, acc);
  store_rowtile<false, false>(acc, nullptr, o0, rowBase, M);
  wmma_rowtile_64(As, WsT[1], rowTile, acc);
  store_rowtile<false, false>(acc, nullptr, o1, rowBase, M);
  wmma_rowtile_64(As, WsT[2], rowTile, acc);
  store_rowtile<false, false>(acc, nullptr, o2, rowBase, M);
}

// ---------------------------------------------------------------------------
// K2: pos-MLP layer 1 (3->64) per edge, plus BN stat accumulation.
// ---------------------------------------------------------------------------
__global__ void __launch_bounds__(256)
pos_lin1_kernel(const float* __restrict__ pos, const int* __restrict__ src,
                const int* __restrict__ dst, int E, int Etot,
                const float* __restrict__ w1, const float* __restrict__ b1,
                float* __restrict__ out, float* statSum, float* statSq) {
  __shared__ float ws[3 * 64], bs[64];
  __shared__ float dpos[256][3];
  __shared__ float ssum[256], ssq[256];

  const int t = threadIdx.x;
  const int base = blockIdx.x * 256;
  if (t < 192) ws[t] = w1[t];
  if (t < 64) bs[t] = b1[t];
  const int e = base + t;
  if (e < Etot) {
    const int s = (e < E) ? src[e] : (e - E);
    const int d = (e < E) ? dst[e] : (e - E);
    dpos[t][0] = pos[d * 3 + 0] - pos[s * 3 + 0];
    dpos[t][1] = pos[d * 3 + 1] - pos[s * 3 + 1];
    dpos[t][2] = pos[d * 3 + 2] - pos[s * 3 + 2];
  }
  __syncthreads();

  const int c = t & 63, g = t >> 6;
  const float w0 = ws[c], w1c = ws[64 + c], w2c = ws[128 + c], bc = bs[c];
  float ls = 0.f, lq = 0.f;
  for (int i = 0; i < 64; ++i) {
    const int el = g * 64 + i;
    const int ge = base + el;
    if (ge < Etot) {
      const float v = dpos[el][0] * w0 + dpos[el][1] * w1c + dpos[el][2] * w2c + bc;
      out[(size_t)ge * 64 + c] = v;
      ls += v;
      lq += v * v;
    }
  }
  ssum[t] = ls;
  ssq[t] = lq;
  __syncthreads();
  if (t < 64) {
    atomicAdd(&statSum[t], ssum[t] + ssum[64 + t] + ssum[128 + t] + ssum[192 + t]);
    atomicAdd(&statSq[t],  ssq[t]  + ssq[64 + t]  + ssq[128 + t]  + ssq[192 + t]);
  }
}

// ---------------------------------------------------------------------------
// K3: fold BN stats into affine scale/shift:  y = x*scale + shift
// ---------------------------------------------------------------------------
__global__ void bn_finalize_kernel(const float* statSum, const float* statSq,
                                   const float* g, const float* be, float cnt,
                                   float* scale, float* shift) {
  const int c = threadIdx.x;
  if (c < 64) {
    const float mu  = statSum[c] / cnt;
    const float var = statSq[c] / cnt - mu * mu;
    const float sc  = g[c] * rsqrtf(var + 1e-5f);
    scale[c] = sc;
    shift[c] = be[c] - mu * sc;
  }
}

// ---------------------------------------------------------------------------
// K4: edge GEMM layer: A = relu(BN(in)) staged as bf16, WMMA, +bias,
// store f32 (maybe in-place), accumulate BN stats of the output.
// ---------------------------------------------------------------------------
__global__ void __launch_bounds__(256)
edge_gemm2_kernel(const float* __restrict__ in, const float* __restrict__ scaleIn,
                  const float* __restrict__ shiftIn, const float* __restrict__ W,
                  const float* __restrict__ b, float* __restrict__ out, int Etot,
                  float* statSum, float* statSq) {
  __shared__ __bf16 As[128 * LDA];
  __shared__ __bf16 WsT[64 * LDWT];
  __shared__ float  bs[64], scs[64], shs[64];
  __shared__ float  ssum[64], ssq[64];

  const int t = threadIdx.x;
  const int base = blockIdx.x * 128;
  stage_weightsT(W, WsT, t);
  if (t < 64) {
    bs[t] = b[t]; scs[t] = scaleIn[t]; shs[t] = shiftIn[t];
    ssum[t] = 0.f; ssq[t] = 0.f;
  }
  __syncthreads();

#pragma unroll
  for (int j = 0; j < 8; ++j) {
    const int idx4 = t + j * 256;
    const int r = idx4 >> 4;
    const int c4 = (idx4 & 15) * 4;
    const int ge = base + r;
    v4bf bvec;
    if (ge < Etot) {
      const float4 v = *(const float4*)(in + (size_t)ge * 64 + c4);
      bvec[0] = (__bf16)fmaxf(v.x * scs[c4 + 0] + shs[c4 + 0], 0.f);
      bvec[1] = (__bf16)fmaxf(v.y * scs[c4 + 1] + shs[c4 + 1], 0.f);
      bvec[2] = (__bf16)fmaxf(v.z * scs[c4 + 2] + shs[c4 + 2], 0.f);
      bvec[3] = (__bf16)fmaxf(v.w * scs[c4 + 3] + shs[c4 + 3], 0.f);
    } else {
      bvec[0] = (__bf16)0.f; bvec[1] = (__bf16)0.f;
      bvec[2] = (__bf16)0.f; bvec[3] = (__bf16)0.f;
    }
    *(v4bf*)(As + r * LDA + c4) = bvec;
  }
  __syncthreads();

  v8f acc[4];
  wmma_rowtile_64(As, WsT, (t >> 5) * 16, acc);
  edge_epilogue(acc, bs, out, base, Etot, ssum, ssq);
  __syncthreads();
  if (t < 64) { atomicAdd(&statSum[t], ssum[t]); atomicAdd(&statSq[t], ssq[t]); }
}

// ---------------------------------------------------------------------------
// K5: attn-MLP layer 1, fused with delta = relu(BN(t2)) writeback and the
// gather A = a_dst[dst] - a_src[src] + delta.  WMMA + bias + stats.
// ---------------------------------------------------------------------------
__global__ void __launch_bounds__(256)
edge_attn1_kernel(const float* __restrict__ t2, const float* __restrict__ scaleIn,
                  const float* __restrict__ shiftIn, float* __restrict__ delta,
                  const float* __restrict__ a_src, const float* __restrict__ a_dst,
                  const int* __restrict__ srcIdx, const int* __restrict__ dstIdx,
                  int E, int Etot, const float* __restrict__ W,
                  const float* __restrict__ b, float* __restrict__ out,
                  float* statSum, float* statSq) {
  __shared__ __bf16 As[128 * LDA];
  __shared__ __bf16 WsT[64 * LDWT];
  __shared__ float  bs[64], scs[64], shs[64];
  __shared__ float  ssum[64], ssq[64];
  __shared__ int    se[128], de[128];

  const int t = threadIdx.x;
  const int base = blockIdx.x * 128;
  stage_weightsT(W, WsT, t);
  if (t < 64) {
    bs[t] = b[t]; scs[t] = scaleIn[t]; shs[t] = shiftIn[t];
    ssum[t] = 0.f; ssq[t] = 0.f;
  }
  if (t < 128) {
    const int ge = base + t;
    if (ge < Etot) {
      se[t] = (ge < E) ? srcIdx[ge] : (ge - E);
      de[t] = (ge < E) ? dstIdx[ge] : (ge - E);
    } else {
      se[t] = 0; de[t] = 0;
    }
  }
  __syncthreads();

#pragma unroll
  for (int j = 0; j < 8; ++j) {
    const int idx4 = t + j * 256;
    const int r = idx4 >> 4;
    const int c4 = (idx4 & 15) * 4;
    const int ge = base + r;
    v4bf bvec;
    if (ge < Etot) {
      const float4 tv = *(const float4*)(t2 + (size_t)ge * 64 + c4);
      float4 dl;
      dl.x = fmaxf(tv.x * scs[c4 + 0] + shs[c4 + 0], 0.f);
      dl.y = fmaxf(tv.y * scs[c4 + 1] + shs[c4 + 1], 0.f);
      dl.z = fmaxf(tv.z * scs[c4 + 2] + shs[c4 + 2], 0.f);
      dl.w = fmaxf(tv.w * scs[c4 + 3] + shs[c4 + 3], 0.f);
      *(float4*)(delta + (size_t)ge * 64 + c4) = dl;  // persist delta (in-place)
      const float4 ad = *(const float4*)(a_dst + (size_t)de[r] * 64 + c4);
      const float4 as = *(const float4*)(a_src + (size_t)se[r] * 64 + c4);
      bvec[0] = (__bf16)(ad.x - as.x + dl.x);
      bvec[1] = (__bf16)(ad.y - as.y + dl.y);
      bvec[2] = (__bf16)(ad.z - as.z + dl.z);
      bvec[3] = (__bf16)(ad.w - as.w + dl.w);
    } else {
      bvec[0] = (__bf16)0.f; bvec[1] = (__bf16)0.f;
      bvec[2] = (__bf16)0.f; bvec[3] = (__bf16)0.f;
    }
    *(v4bf*)(As + r * LDA + c4) = bvec;
  }
  __syncthreads();

  v8f acc[4];
  wmma_rowtile_64(As, WsT, (t >> 5) * 16, acc);
  edge_epilogue(acc, bs, out, base, Etot, ssum, ssq);
  __syncthreads();
  if (t < 64) { atomicAdd(&statSum[t], ssum[t]); atomicAdd(&statSq[t], ssq[t]); }
}

// ---------------------------------------------------------------------------
// Segment softmax helpers: monotone int encoding of floats for atomicMax.
// ---------------------------------------------------------------------------
__device__ inline int enc_f(float f) {
  int i = __float_as_int(f);
  return (i < 0) ? (i ^ 0x7FFFFFFF) : i;
}
__device__ inline float dec_f(int i) {
  if (i < 0) i ^= 0x7FFFFFFF;
  return __int_as_float(i);
}
__device__ inline float alpha_of(const float* u2, const float* sc,
                                 const float* sh, int idx, int c) {
  return fmaxf(u2[idx] * sc[c] + sh[c], 0.f);
}

__global__ void __launch_bounds__(256)
smax_max_kernel(const float* __restrict__ u2, const float* __restrict__ sc,
                const float* __restrict__ sh, const int* __restrict__ dst,
                int E, int Etot, int* __restrict__ nodemax) {
  const int idx = blockIdx.x * 256 + threadIdx.x;
  const int e = idx >> 6, c = idx & 63;
  if (e >= Etot) return;
  const int d = (e < E) ? dst[e] : (e - E);
  atomicMax(&nodemax[d * 64 + c], enc_f(alpha_of(u2, sc, sh, idx, c)));
}

__global__ void __launch_bounds__(256)
smax_sum_kernel(const float* __restrict__ u2, const float* __restrict__ sc,
                const float* __restrict__ sh, const int* __restrict__ dst,
                int E, int Etot, const int* __restrict__ nodemax,
                float* __restrict__ nodesum) {
  const int idx = blockIdx.x * 256 + threadIdx.x;
  const int e = idx >> 6, c = idx & 63;
  if (e >= Etot) return;
  const int d = (e < E) ? dst[e] : (e - E);
  const float m = dec_f(nodemax[d * 64 + c]);
  atomicAdd(&nodesum[d * 64 + c], __expf(alpha_of(u2, sc, sh, idx, c) - m));
}

__global__ void __launch_bounds__(256)
aggregate_kernel(const float* __restrict__ u2, const float* __restrict__ sc,
                 const float* __restrict__ sh, const int* __restrict__ src,
                 const int* __restrict__ dst, int E, int Etot,
                 const int* __restrict__ nodemax, const float* __restrict__ nodesum,
                 const float* __restrict__ vfeat, const float* __restrict__ delta,
                 float* __restrict__ out_pre) {
  const int idx = blockIdx.x * 256 + threadIdx.x;
  const int e = idx >> 6, c = idx & 63;
  if (e >= Etot) return;
  const int s = (e < E) ? src[e] : (e - E);
  const int d = (e < E) ? dst[e] : (e - E);
  const float m = dec_f(nodemax[d * 64 + c]);
  const float ssumv = nodesum[d * 64 + c];
  const float coeff = __expf(alpha_of(u2, sc, sh, idx, c) - m) / (ssumv + 1e-16f);
  const float val = coeff * (vfeat[(size_t)s * 64 + c] + delta[idx]);
  atomicAdd(&out_pre[d * 64 + c], val);
}

// ---------------------------------------------------------------------------
// Host-side launch
// ---------------------------------------------------------------------------
extern "C" void kernel_launch(void* const* d_in, const int* in_sizes, int n_in,
                              void* d_out, int out_size, void* d_ws, size_t ws_size,
                              hipStream_t stream) {
  const float* x       = (const float*)d_in[0];
  const float* pos     = (const float*)d_in[1];
  const float* w_in    = (const float*)d_in[2];
  const float* b_in    = (const float*)d_in[3];
  const float* w_lin   = (const float*)d_in[4];
  const float* w_src   = (const float*)d_in[5];
  const float* w_dst   = (const float*)d_in[6];
  const float* pos_w1  = (const float*)d_in[7];
  const float* pos_b1  = (const float*)d_in[8];
  const float* pos_g1  = (const float*)d_in[9];
  const float* pos_be1 = (const float*)d_in[10];
  const float* pos_w2  = (const float*)d_in[11];
  const float* pos_b2  = (const float*)d_in[12];
  const float* pos_g2  = (const float*)d_in[13];
  const float* pos_be2 = (const float*)d_in[14];
  const float* attn_w1 = (const float*)d_in[15];
  const float* attn_b1 = (const float*)d_in[16];
  const float* attn_g1 = (const float*)d_in[17];
  const float* attn_be1= (const float*)d_in[18];
  const float* attn_w2 = (const float*)d_in[19];
  const float* attn_b2 = (const float*)d_in[20];
  const float* attn_g2 = (const float*)d_in[21];
  const float* attn_be2= (const float*)d_in[22];
  const float* w_out   = (const float*)d_in[23];
  const float* b_out   = (const float*)d_in[24];
  const int*   eidx    = (const int*)d_in[25];

  const int N    = in_sizes[0] / 64;
  const int E    = in_sizes[25] / 2;
  const int Etot = E + N;
  const int* srcI = eidx;
  const int* dstI = eidx + E;

  // -------- workspace carve-up (float elements) --------
  float* ws = (float*)d_ws;
  size_t off = 0;
  float* bufA   = ws + off; off += (size_t)Etot * 64;  // t1 -> u1 -> u2 (in-place)
  float* bufB   = ws + off; off += (size_t)Etot * 64;  // t2 -> delta (in-place)
  float* h      = ws + off; off += (size_t)N * 64;
  float* a_srcB = ws + off; off += (size_t)N * 64;
  float* a_dstB = ws + off; off += (size_t)N * 64;
  float* vfeat  = ws + off; off += (size_t)N * 64;
  float* outPre = ws + off; off += (size_t)N * 64;
  int*   nmax   = (int*)(ws + off); off += (size_t)N * 64;
  float* nsum   = ws + off; off += (size_t)N * 64;
  float* stats  = ws + off; off += 512;  // 4 layers x (sum[64], sumsq[64])
  float* bnp    = ws + off; off += 512;  // 4 layers x (scale[64], shift[64])

  float* sSum[4], *sSq[4], *bScale[4], *bShift[4];
  for (int l = 0; l < 4; ++l) {
    sSum[l]   = stats + l * 128;
    sSq[l]    = stats + l * 128 + 64;
    bScale[l] = bnp   + l * 128;
    bShift[l] = bnp   + l * 128 + 64;
  }

  const int Nc = N * 64;
  const float cnt = (float)Etot;
  const int gNode = (N + 127) / 128;
  const int gEdge128 = (Etot + 127) / 128;
  const int gEdge256 = (Etot + 255) / 256;
  const int gElem = (Etot * 64 + 255) / 256;

  // K0: init accumulators
  init_kernel<<<(Nc + 255) / 256, 256, 0, stream>>>(nsum, nmax, outPre, stats, Nc);

  // K1: node GEMM + fused projections (WMMA bf16)
  gemm64_kernel<true, true><<<gNode, 256, 0, stream>>>(x, w_in, b_in, h, N);
  proj3_kernel<<<gNode, 256, 0, stream>>>(h, w_src, w_dst, w_lin,
                                          a_srcB, a_dstB, vfeat, N);

  // K2: pos MLP layer 1 (3->64) + stats L0
  pos_lin1_kernel<<<gEdge256, 256, 0, stream>>>(pos, srcI, dstI, E, Etot,
                                                pos_w1, pos_b1, bufA, sSum[0], sSq[0]);
  bn_finalize_kernel<<<1, 64, 0, stream>>>(sSum[0], sSq[0], pos_g1, pos_be1, cnt,
                                           bScale[0], bShift[0]);

  // K4: pos MLP layer 2 (WMMA) : t2 = bn_relu(t1) @ pos_w2 + b2, stats L1
  edge_gemm2_kernel<<<gEdge128, 256, 0, stream>>>(bufA, bScale[0], bShift[0],
                                                  pos_w2, pos_b2, bufB, Etot,
                                                  sSum[1], sSq[1]);
  bn_finalize_kernel<<<1, 64, 0, stream>>>(sSum[1], sSq[1], pos_g2, pos_be2, cnt,
                                           bScale[1], bShift[1]);

  // K5: delta writeback + attn layer 1 (WMMA), stats L2
  edge_attn1_kernel<<<gEdge128, 256, 0, stream>>>(bufB, bScale[1], bShift[1], bufB,
                                                  a_srcB, a_dstB, srcI, dstI, E, Etot,
                                                  attn_w1, attn_b1, bufA,
                                                  sSum[2], sSq[2]);
  bn_finalize_kernel<<<1, 64, 0, stream>>>(sSum[2], sSq[2], attn_g1, attn_be1, cnt,
                                           bScale[2], bShift[2]);

  // K4: attn layer 2 (WMMA, in-place), stats L3
  edge_gemm2_kernel<<<gEdge128, 256, 0, stream>>>(bufA, bScale[2], bShift[2],
                                                  attn_w2, attn_b2, bufA, Etot,
                                                  sSum[3], sSq[3]);
  bn_finalize_kernel<<<1, 64, 0, stream>>>(sSum[3], sSq[3], attn_g2, attn_be2, cnt,
                                           bScale[3], bShift[3]);

  // segment softmax + weighted aggregation (alpha recomputed from u2 each pass)
  smax_max_kernel<<<gElem, 256, 0, stream>>>(bufA, bScale[3], bShift[3], dstI,
                                             E, Etot, nmax);
  smax_sum_kernel<<<gElem, 256, 0, stream>>>(bufA, bScale[3], bShift[3], dstI,
                                             E, Etot, nmax, nsum);
  aggregate_kernel<<<gElem, 256, 0, stream>>>(bufA, bScale[3], bShift[3], srcI, dstI,
                                              E, Etot, nmax, nsum, vfeat, bufB, outPre);

  // final node GEMM: out = relu(out_pre @ w_out + b_out)
  gemm64_kernel<true, true><<<gNode, 256, 0, stream>>>(outPre, w_out, b_out,
                                                       (float*)d_out, N);
}
